// GIN_53661321396793
// MI455X (gfx1250) — matrix operand order repaired and checked by reading, not compiled, not run on hardware
//
#include <hip/hip_runtime.h>
#include <math.h>

// ---------------------------------------------------------------------------
// GIN embedding for MI455X (gfx1250, wave32).
//   - BatchNorm (batch stats) -> 5 x [scatter-sum + GEMM(128x128) + tanh] -> final GEMM
//   - GEMMs use V_WMMA_F32_16X16X32_BF16 with split-bf16 (hi+lo) operands for
//     ~fp32 accuracy; accumulate in fp32.
//   - Edge aggregation via global f32 atomics (x array is L2-resident: 25.6MB).
// ---------------------------------------------------------------------------

typedef __bf16 bf16_t;
typedef __attribute__((ext_vector_type(16))) __bf16 v16bf;
typedef __attribute__((ext_vector_type(8)))  float  v8f;

static constexpr int   DIM    = 128;     // feature dim (C == D == 128)
static constexpr float BN_EPS = 1e-5f;

// ---------------------------------------------------------------------- utils
__global__ void zero_kernel(float* p, int n) {
    int i = blockIdx.x * blockDim.x + threadIdx.x;
    if (i < n) p[i] = 0.0f;
}

__global__ void copy_kernel(float4* __restrict__ dst, const float4* __restrict__ src, int n4) {
    int i = blockIdx.x * blockDim.x + threadIdx.x;
    int stride = gridDim.x * blockDim.x;
    for (; i < n4; i += stride) dst[i] = src[i];
}

// ------------------------------------------------------------------ batchnorm
// Per-block partial sums over 128 rows; one atomic per column per block.
__global__ void bn_stats_kernel(const float* __restrict__ X, const float* __restrict__ imp,
                                float* __restrict__ stats /*[256]*/, int N) {
    int col  = threadIdx.x;                 // 0..127
    int row0 = blockIdx.x * 128;
    int rend = row0 + 128; if (rend > N) rend = N;
    float s = 0.0f, s2 = 0.0f;
    for (int r = row0; r < rend; ++r) {
        float v = X[(size_t)r * DIM + col] * imp[r];
        s += v; s2 += v * v;
    }
    atomicAdd(&stats[col], s);
    atomicAdd(&stats[DIM + col], s2);
}

__global__ void bn_finalize_kernel(const float* __restrict__ stats,
                                   const float* __restrict__ gamma,
                                   const float* __restrict__ beta,
                                   float* __restrict__ ss /*[256] scale,shift*/, int N) {
    int col = threadIdx.x;                  // 0..127
    float invn = 1.0f / (float)N;
    float mean = stats[col] * invn;
    float var  = stats[DIM + col] * invn - mean * mean;   // ddof=0, matches jnp.var
    float sc   = gamma[col] * rsqrtf(var + BN_EPS);
    ss[col]        = sc;
    ss[DIM + col]  = beta[col] - mean * sc;
}

// y = (X*imp)*scale + shift; write into xcur and h (h is the layer-0 self term).
__global__ void bn_apply_kernel(const float* __restrict__ X, const float* __restrict__ imp,
                                const float* __restrict__ ss,
                                float* __restrict__ xcur, float* __restrict__ hbuf, int total) {
    int i = blockIdx.x * blockDim.x + threadIdx.x;
    if (i >= total) return;
    int col = i & (DIM - 1);
    int row = i >> 7;
    float v = X[i] * imp[row];
    v = v * ss[col] + ss[DIM + col];
    xcur[i] = v;
    hbuf[i] = v;
}

// ---------------------------------------------------------------- aggregation
// One edge per 32-lane wave; each lane handles 4 columns (float4 gather + 4 atomics).
__global__ void edge_agg_kernel(const float* __restrict__ x, float* __restrict__ hbuf,
                                const int* __restrict__ ei /*[2,E]*/, int E) {
    int tid  = blockIdx.x * blockDim.x + threadIdx.x;
    int e    = tid >> 5;
    if (e >= E) return;
    int c4   = (tid & 31) << 2;             // column base 0..124
    int s    = ei[e];                       // src = edge_index[0][e]
    int d    = ei[E + e];                   // dst = edge_index[1][e]
    float4 v = *(const float4*)(x + (size_t)s * DIM + c4);
    float* dst = hbuf + (size_t)d * DIM + c4;
    atomicAdd(dst + 0, v.x);
    atomicAdd(dst + 1, v.y);
    atomicAdd(dst + 2, v.z);
    atomicAdd(dst + 3, v.w);
}

// ----------------------------------------------------- weight fragment packer
// Pack W[128x128] (row-major, W[k][n]) into per-lane contiguous B fragments for
// V_WMMA_*_16x16x32 (16-bit B 32x16 layout): lanes 0-15 -> n, K=0..7,16..23 across the
// 16 half slots; lanes 16-31 -> same n, K=8..15,24..31.  frag idx = ((kb*8+nt)*32+lane)*16+j.
__global__ void wconvert_kernel(const float* __restrict__ W,
                                bf16_t* __restrict__ hi, bf16_t* __restrict__ lo) {
    int idx = blockIdx.x * blockDim.x + threadIdx.x;
    if (idx >= 4 * 8 * 32 * 16) return;
    int j    = idx & 15;
    int lane = (idx >> 4) & 31;
    int nt   = (idx >> 9) & 7;
    int kb   = idx >> 12;
    int n    = nt * 16 + (lane & 15);
    int k    = kb * 32 + (lane >> 4) * 8 + (j & 7) + (j >> 3) * 16;
    float  v = W[(size_t)k * DIM + n];
    bf16_t h = (bf16_t)v;
    hi[idx]  = h;
    lo[idx]  = (bf16_t)(v - (float)h);
}

// -------------------------------------------------------------- WMMA GEMM+tanh
// One wave per 16-row M tile; loops the 8 N-tiles keeping A fragments resident.
// Split-bf16: acc += Ahi*Bhi + Ahi*Blo + Alo*Bhi  (fp32 accumulate).
__global__ __launch_bounds__(256)
void gemm_tanh_kernel(const float* __restrict__ Hin,
                      const bf16_t* __restrict__ Whi, const bf16_t* __restrict__ Wlo,
                      const float* __restrict__ bias,   // may be null (final layer)
                      float* __restrict__ Xout,         // may be null (final layer)
                      float* __restrict__ Out,          // out + g*N*768 + slot*128
                      int N) {
    int lane = threadIdx.x & 31;
    int wv   = threadIdx.x >> 5;
    int gw   = blockIdx.x * 8 + wv;
    int m0   = gw * 16;
    if (m0 >= N) return;                    // wave-uniform: EXEC stays full
    int hs = lane >> 4;                     // K-half select
    int ml = lane & 15;                     // M row within tile / N col within tile
    int m  = m0 + ml; if (m >= N) m = N - 1;   // clamp loads; stores predicated below
    const float* row = Hin + (size_t)m * DIM;

    // A fragments for all 4 k-blocks (16-bit A 16x32 layout):
    //   j in 0..7  -> K = kb*32 + hs*8 + j
    //   j in 8..15 -> K = kb*32 + 16 + hs*8 + (j-8)
    v16bf ahi[4], alo[4];
#pragma unroll
    for (int kb = 0; kb < 4; ++kb) {
        int kbase = kb * 32 + hs * 8;
#pragma unroll
        for (int j = 0; j < 8; ++j) {
            float v0 = row[kbase + j];
            float v1 = row[kbase + 16 + j];
            bf16_t h0 = (bf16_t)v0, h1 = (bf16_t)v1;
            ahi[kb][j]     = h0;  alo[kb][j]     = (bf16_t)(v0 - (float)h0);
            ahi[kb][8 + j] = h1;  alo[kb][8 + j] = (bf16_t)(v1 - (float)h1);
        }
    }

#pragma unroll
    for (int nt = 0; nt < 8; ++nt) {
        v8f acc = {0.f, 0.f, 0.f, 0.f, 0.f, 0.f, 0.f, 0.f};
#pragma unroll
        for (int kb = 0; kb < 4; ++kb) {
            const bf16_t* bp = Whi + ((size_t)((kb * 8 + nt) * 32 + lane) * 16);
            const bf16_t* bq = Wlo + ((size_t)((kb * 8 + nt) * 32 + lane) * 16);
            v16bf bhi, blo;
#pragma unroll
            for (int j = 0; j < 16; ++j) { bhi[j] = bp[j]; blo[j] = bq[j]; }
            acc = __builtin_amdgcn_wmma_f32_16x16x32_bf16(false, ahi[kb], false, bhi,
                                                          (short)0, acc, false, false);
            acc = __builtin_amdgcn_wmma_f32_16x16x32_bf16(false, ahi[kb], false, blo,
                                                          (short)0, acc, false, false);
            acc = __builtin_amdgcn_wmma_f32_16x16x32_bf16(false, alo[kb], false, bhi,
                                                          (short)0, acc, false, false);
        }
        int ncol = nt * 16 + ml;
        float bv = bias ? bias[ncol] : 0.0f;
        // C/D layout: lane L -> n = L&15 ; VGPR v -> m = m0 + v + 8*(L>>4)
#pragma unroll
        for (int v = 0; v < 8; ++v) {
            int mm = m0 + v + 8 * hs;
            if (mm < N) {
                float val = tanhf(acc[v] + bv);
                if (Xout) Xout[(size_t)mm * DIM + ncol] = val;
                Out[(size_t)mm * 768 + ncol] = val;
            }
        }
    }
}

// ------------------------------------------------------------------- launcher
static inline size_t align256(size_t x) { return (x + 255) & ~(size_t)255; }

extern "C" void kernel_launch(void* const* d_in, const int* in_sizes, int n_in,
                              void* d_out, int out_size, void* d_ws, size_t ws_size,
                              hipStream_t stream) {
    const float* src_x   = (const float*)d_in[0];
    const float* src_imp = (const float*)d_in[1];
    const float* tgt_x   = (const float*)d_in[2];
    const float* tgt_imp = (const float*)d_in[3];
    const float* gamma   = (const float*)d_in[4];
    const float* beta    = (const float*)d_in[5];
    const float* Wmat[6] = { (const float*)d_in[6],  (const float*)d_in[8],
                             (const float*)d_in[10], (const float*)d_in[12],
                             (const float*)d_in[14], (const float*)d_in[16] };
    const float* bvec[5] = { (const float*)d_in[7],  (const float*)d_in[9],
                             (const float*)d_in[11], (const float*)d_in[13],
                             (const float*)d_in[15] };
    const int* ei_s = (const int*)d_in[17];
    const int* ei_t = (const int*)d_in[18];

    const int N = in_sizes[1];            // src_imp is [N,1]
    const int E = in_sizes[17] / 2;       // edge_index is [2,E]
    float* out = (float*)d_out;

    // workspace layout
    char* ws = (char*)d_ws;
    size_t xbytes = align256((size_t)N * DIM * sizeof(float));
    float* buf0 = (float*)ws;               ws += xbytes;   // x_cur / x_next
    float* buf1 = (float*)ws;               ws += xbytes;   // h (self + agg)
    float* buf2 = (float*)ws;               ws += xbytes;   // x_next / x_cur
    float* stats = (float*)ws;              ws += align256(256 * sizeof(float));
    float* ss    = (float*)ws;              ws += align256(256 * sizeof(float));
    const size_t fragElems = 4 * 8 * 32 * 16;               // 16384 bf16 per part
    bf16_t* whi[6]; bf16_t* wlo[6];
    for (int l = 0; l < 6; ++l) {
        whi[l] = (bf16_t*)ws; ws += align256(fragElems * sizeof(bf16_t));
        wlo[l] = (bf16_t*)ws; ws += align256(fragElems * sizeof(bf16_t));
    }

    // pack weights into WMMA B-fragment layout (hi/lo split-bf16)
    for (int l = 0; l < 6; ++l)
        wconvert_kernel<<<(int)((fragElems + 255) / 256), 256, 0, stream>>>(Wmat[l], whi[l], wlo[l]);

    const int total     = N * DIM;
    const int elemBlks  = (total + 255) / 256;
    const int statBlks  = (N + 127) / 128;
    const int edgeBlks  = (int)(((long long)E * 32 + 255) / 256);
    const int gemmBlks  = (N + 127) / 128;                  // 8 waves x 16 rows per block
    const int n4        = total / 4;

    for (int g = 0; g < 2; ++g) {
        const float* X   = g ? tgt_x   : src_x;
        const float* IMP = g ? tgt_imp : src_imp;
        const int*   EI  = g ? ei_t    : ei_s;
        float* outg = out + (size_t)g * N * 768;

        zero_kernel<<<1, 256, 0, stream>>>(stats, 256);
        bn_stats_kernel<<<statBlks, 128, 0, stream>>>(X, IMP, stats, N);
        bn_finalize_kernel<<<1, 128, 0, stream>>>(stats, gamma, beta, ss, N);
        bn_apply_kernel<<<elemBlks, 256, 0, stream>>>(X, IMP, ss, buf0, buf1, total);

        float* xcur  = buf0;
        float* hbuf  = buf1;
        float* xnext = buf2;
        for (int l = 0; l < 5; ++l) {
            if (l > 0)
                copy_kernel<<<4096, 256, 0, stream>>>((float4*)hbuf, (const float4*)xcur, n4);
            edge_agg_kernel<<<edgeBlks, 256, 0, stream>>>(xcur, hbuf, EI, E);
            gemm_tanh_kernel<<<gemmBlks, 256, 0, stream>>>(hbuf, whi[l], wlo[l], bvec[l],
                                                           xnext, outg + (size_t)l * DIM, N);
            float* t = xcur; xcur = xnext; xnext = t;
        }
        // final: tanh(x5 @ Wf), no bias, output slot 5 only
        gemm_tanh_kernel<<<gemmBlks, 256, 0, stream>>>(xcur, whi[5], wlo[5], nullptr,
                                                       nullptr, outg + (size_t)5 * DIM, N);
    }
}
